// MyGat_34059090657639
// MI455X (gfx1250) — matrix operand order repaired and checked by reading, not compiled, not run on hardware
//
#include <hip/hip_runtime.h>
#include <hip/hip_bf16.h>

#define N_NODES 8192
#define F_IN    512
#define F_OUT   256

typedef __attribute__((ext_vector_type(16))) __bf16 v16bf;
typedef __attribute__((ext_vector_type(8)))  float  v8f;

// fp32 -> bf16 (round to nearest even), returned as raw bits
__device__ __forceinline__ unsigned short f2bf(float f) {
    unsigned u = __float_as_uint(f);
    unsigned r = u + 0x7FFFu + ((u >> 16) & 1u);
    return (unsigned short)(r >> 16);
}

// 16-bit A-fragment (16x32, MxK) lane/element mapping used here: contiguous k = 16h + e
__device__ __forceinline__ int a_idx(int m, int k) {
    int h    = (k >> 4) & 1;                 // lane half
    int lane = m + (h << 4);
    int e    = k & 15;                       // element 0..15 within lane
    return lane * 16 + e;
}
// CDNA5 16-bit B-fragment (32x16, KxN)
__device__ __forceinline__ int b_idx(int k, int n) {
    int lane = n + ((k >> 4) << 4);
    int e    = k & 15;
    return lane * 16 + e;
}

// ---------------------------------------------------------------------------
// Phase A: H_prime = x @ W^T (bf16 WMMA, fp32 accum), alpha_src/dst reductions.
// H_prime is written to workspace ALREADY in B-fragment tile order:
//   hp_frag[((jc*16 + ftile)*32 + lane)*16 + e],  jc = row>>5
// so phase B can stream it to LDS with zero layout shuffling.
// ---------------------------------------------------------------------------
__global__ __launch_bounds__(256)
void gat_phaseA(const float* __restrict__ x, const float* __restrict__ w,
                const float* __restrict__ att_src, const float* __restrict__ att_dst,
                unsigned short* __restrict__ hp_frag,
                float* __restrict__ asrc_g, float* __restrict__ adst_g) {
    __shared__ __align__(32) unsigned short a_stage[2 * 512];   // 2 i-tiles
    __shared__ __align__(32) unsigned short b_stage[16 * 512];  // 16 f-tiles
    __shared__ float lds_asrc[32];
    __shared__ float lds_adst[32];

    const int tid   = threadIdx.x;
    const int lane  = tid & 31;
    const int wid   = tid >> 5;
    const int itile = wid & 1;      // which 16-row tile
    const int fgrp  = wid >> 1;     // 4 f-tiles per wave
    const int i0    = blockIdx.x * 32;

    if (tid < 32) { lds_asrc[tid] = 0.0f; lds_adst[tid] = 0.0f; }

    v8f acc[4] = {};

    const int il = tid >> 3;          // 0..31 : x row within block
    const int kq = (tid & 7) << 2;    // 0..28 : k sub-base

    for (int k0 = 0; k0 < F_IN; k0 += 32) {
        // ---- stage x tile (32x32 fp32 -> bf16, A-fragment order) ----
        float4 xv = *(const float4*)(x + (size_t)(i0 + il) * F_IN + k0 + kq);
        const float xa[4] = {xv.x, xv.y, xv.z, xv.w};
#pragma unroll
        for (int q = 0; q < 4; ++q) {
            int kl = kq + q;
            a_stage[((il >> 4) << 9) + a_idx(il & 15, kl)] = f2bf(xa[q]);
        }
        // ---- stage W tile: B[k][f] = w[f][k], thread t owns f = t ----
        const float4* wr = (const float4*)(w + (size_t)tid * F_IN + k0);
#pragma unroll
        for (int p4 = 0; p4 < 8; ++p4) {
            float4 wv = wr[p4];
            const float wa[4] = {wv.x, wv.y, wv.z, wv.w};
#pragma unroll
            for (int q = 0; q < 4; ++q) {
                int kl = (p4 << 2) + q;
                b_stage[((tid >> 4) << 9) + b_idx(kl, tid & 15)] = f2bf(wa[q]);
            }
        }
        __syncthreads();
        // ---- WMMA: each wave does its i-tile x 4 f-tiles ----
        v16bf a = *(const v16bf*)(a_stage + (itile << 9) + lane * 16);
#pragma unroll
        for (int t = 0; t < 4; ++t) {
            v16bf b = *(const v16bf*)(b_stage + ((fgrp * 4 + t) << 9) + lane * 16);
            acc[t] = __builtin_amdgcn_wmma_f32_16x16x32_bf16(
                false, a, false, b, (short)0, acc[t], false, false);
        }
        __syncthreads();
    }

    // ---- epilogue: fragment-ordered bf16 H_prime + alpha reductions ----
    const int n = lane & 15, h = lane >> 4;
    float psrc[8], pdst[8];
#pragma unroll
    for (int v = 0; v < 8; ++v) { psrc[v] = 0.0f; pdst[v] = 0.0f; }
#pragma unroll
    for (int t = 0; t < 4; ++t) {
        int   f  = fgrp * 64 + t * 16 + n;
        float as = att_src[f];
        float ad = att_dst[f];
        unsigned pk[4];
#pragma unroll
        for (int p = 0; p < 4; ++p) {
            pk[p] = (unsigned)f2bf(acc[t][2 * p]) |
                    ((unsigned)f2bf(acc[t][2 * p + 1]) << 16);
        }
#pragma unroll
        for (int v = 0; v < 8; ++v) {
            psrc[v] += acc[t][v] * as;
            pdst[v] += acc[t][v] * ad;
        }
        // rows v=0..7 map to consecutive fragment elements e=8h..8h+7:
        // one 16-byte store covers the whole run.
        size_t base = ((size_t)(blockIdx.x * 16 + fgrp * 4 + t) * 32 +
                       (n + 16 * itile)) * 16 + 8 * h;
        *(uint4*)(hp_frag + base) = make_uint4(pk[0], pk[1], pk[2], pk[3]);
    }
#pragma unroll
    for (int v = 0; v < 8; ++v) {
        int rl = itile * 16 + v + 8 * h;
        atomicAdd(&lds_asrc[rl], psrc[v]);
        atomicAdd(&lds_adst[rl], pdst[v]);
    }
    __syncthreads();
    if (tid < 32) {
        asrc_g[i0 + tid] = lds_asrc[tid];
        adst_g[i0 + tid] = lds_adst[tid];
    }
}

// ---------------------------------------------------------------------------
// Phase B: fused attention. Streams adj 32x32 tiles; Tau tiles built in
// A-fragment order via v_exp_f32; H_prime fragments pulled into LDS with
// GLOBAL_LOAD_ASYNC_TO_LDS_B128 (ASYNCcnt); Z accumulated with bf16 WMMA.
// ---------------------------------------------------------------------------
__global__ __launch_bounds__(256)
void gat_phaseB(const float* __restrict__ adj,
                const unsigned short* __restrict__ hp_frag,
                const float* __restrict__ asrc_g, const float* __restrict__ adst_g,
                const float* __restrict__ bias, float* __restrict__ out) {
    __shared__ __align__(32) unsigned short a_stage[2 * 512];
    __shared__ __align__(32) unsigned short b_stage[16 * 512];
    __shared__ float lds_adst[32];
    __shared__ float lds_rs[32];

    const int tid   = threadIdx.x;
    const int lane  = tid & 31;
    const int wid   = tid >> 5;
    const int itile = wid & 1;
    const int fgrp  = wid >> 1;
    const int i0    = blockIdx.x * 32;

    if (tid < 32) { lds_adst[tid] = adst_g[i0 + tid]; lds_rs[tid] = 0.0f; }
    __syncthreads();

    v8f   acc[4] = {};
    float rs[4]  = {0.0f, 0.0f, 0.0f, 0.0f};

    const int jl = tid >> 3;          // 0..31 : j within chunk
    const int ib = (tid & 7) << 2;    // 0..28 : i sub-base (4 consecutive i)

    // LDS byte address of this thread's 64B slice of b_stage (addr[31:0] is
    // the LDS offset in the shared aperture, ISA 10.2)
    const unsigned lds_b = (unsigned)(size_t)(void*)b_stage + (unsigned)(tid * 64);
    const unsigned voff  = (unsigned)(tid * 64);   // matching global byte offset

    for (int j0 = 0; j0 < N_NODES; j0 += 32) {
        // ---- async-copy this chunk's H_prime fragments (16KB) into LDS ----
        const unsigned short* gsrc = hp_frag + (size_t)(j0 >> 5) * (16 * 512);
        asm volatile(
            "global_load_async_to_lds_b128 %0, %1, %2 offset:0\n\t"
            "global_load_async_to_lds_b128 %0, %1, %2 offset:16\n\t"
            "global_load_async_to_lds_b128 %0, %1, %2 offset:32\n\t"
            "global_load_async_to_lds_b128 %0, %1, %2 offset:48"
            :: "v"(lds_b), "v"(voff), "s"(gsrc) : "memory");

        // ---- adj tile (transposed access: adj[j, i]) + scores ----
        float4 av = *(const float4*)(adj + (size_t)(j0 + jl) * N_NODES + i0 + ib);
        if (j0 + 32 < N_NODES)   // prefetch next adj tile (global_prefetch_b8)
            __builtin_prefetch(adj + (size_t)(j0 + 32 + jl) * N_NODES + i0 + ib, 0, 1);
        float cj = asrc_g[j0 + jl];
        const float aa[4] = {av.x, av.y, av.z, av.w};
#pragma unroll
        for (int q = 0; q < 4; ++q) {
            int   iloc = ib + q;
            float c    = cj + lds_adst[iloc];
            float tau  = aa[q] * __expf(fmaxf(0.2f * c, c));
            rs[q] += tau;
            a_stage[((iloc >> 4) << 9) + a_idx(iloc & 15, jl)] = f2bf(tau);
        }

        asm volatile("s_wait_asynccnt 0x0" ::: "memory");
        __syncthreads();

        // ---- WMMA: Z_unnorm += Tau_tile @ Hp_tile ----
        v16bf a = *(const v16bf*)(a_stage + (itile << 9) + lane * 16);
#pragma unroll
        for (int t = 0; t < 4; ++t) {
            v16bf b = *(const v16bf*)(b_stage + ((fgrp * 4 + t) << 9) + lane * 16);
            acc[t] = __builtin_amdgcn_wmma_f32_16x16x32_bf16(
                false, a, false, b, (short)0, acc[t], false, false);
        }
        __syncthreads();
    }

    // ---- row-sum reduction ----
#pragma unroll
    for (int q = 0; q < 4; ++q) atomicAdd(&lds_rs[ib + q], rs[q]);
    __syncthreads();

    // ---- normalize + bias, write fp32 output ----
    const int n = lane & 15, h = lane >> 4;
#pragma unroll
    for (int t = 0; t < 4; ++t) {
        int   f  = fgrp * 64 + t * 16 + n;
        float bv = bias[f];
#pragma unroll
        for (int v = 0; v < 8; ++v) {
            int rl = itile * 16 + v + 8 * h;
            out[(size_t)(i0 + rl) * F_OUT + f] = acc[t][v] / lds_rs[rl] + bv;
        }
    }
}

// ---------------------------------------------------------------------------
extern "C" void kernel_launch(void* const* d_in, const int* in_sizes, int n_in,
                              void* d_out, int out_size, void* d_ws, size_t ws_size,
                              hipStream_t stream) {
    const float* x    = (const float*)d_in[0];
    const float* adj  = (const float*)d_in[1];
    const float* wgt  = (const float*)d_in[2];
    const float* asv  = (const float*)d_in[3];
    const float* adv  = (const float*)d_in[4];
    const float* bias = (const float*)d_in[5];
    float*       out  = (float*)d_out;

    // workspace: [ H_prime (B-fragment order) bf16 : N*F_OUT*2 B ]
    //            [ alpha_src : N f32 ][ alpha_dst : N f32 ]
    unsigned short* hp   = (unsigned short*)d_ws;
    float*          asrc = (float*)((char*)d_ws + (size_t)N_NODES * F_OUT * sizeof(unsigned short));
    float*          adst = asrc + N_NODES;

    dim3 grid(N_NODES / 32), block(256);
    hipLaunchKernelGGL(gat_phaseA, grid, block, 0, stream,
                       x, wgt, asv, adv, hp, asrc, adst);
    hipLaunchKernelGGL(gat_phaseB, grid, block, 0, stream,
                       adj, hp, asrc, adst, bias, out);
}
